// gat_1_10892037063088
// MI455X (gfx1250) — compile-verified, matrix operand order checked by reading, not run on hardware
//
#include <hip/hip_runtime.h>
#include <hip/hip_bf16.h>
#include <math.h>

typedef __attribute__((ext_vector_type(16))) _Float16 v16h;
typedef __attribute__((ext_vector_type(8)))  _Float16 v8h;
typedef __attribute__((ext_vector_type(8)))  float    v8f;

#define N_NODES 30000
#define N_EDGES 480000
#define E_TOT   (N_EDGES + N_NODES)
#define HEADS   8

// ---------------------------------------------------------------------------
// WMMA GEMM:  G[M,N] = A16[M,K] @ Wt16[N,K]^T   (f16 in, f32 accumulate/out)
// block = 256 threads (8 waves), tile 128x64, K-step 32.
// A16 row-major, Wt16 transposed (N x K) so both LDS tiles stage with
// contiguous 16B chunks: global_load_b128 -> ds_store_b128, and fragments
// load as two ds_load_b128 each. Register double-buffer hides global latency.
// ---------------------------------------------------------------------------
#define BM 128
#define BN 64
#define BK 32
#define LDA  40   // halfs per A row    (multiple of 8 -> 16B aligned rows)
#define LDBT 40   // halfs per B column (transposed layout)

__global__ __launch_bounds__(256)
void gemm_f16_wmma(const _Float16* __restrict__ A, const _Float16* __restrict__ Wt,
                   float* __restrict__ G, int M, int K, int N)
{
    __shared__ __align__(16) _Float16 As [BM * LDA];   // [row][k]
    __shared__ __align__(16) _Float16 BsT[BN * LDBT];  // [col][k]

    const int tid  = threadIdx.x;
    const int wave = tid >> 5;
    const int lane = tid & 31;
    const int m0 = blockIdx.y * BM;
    const int n0 = blockIdx.x * BN;

    union F16x16 { v16h v; v8h h8[2]; };
    union F32x8  { v8f  v; float f[8]; };

    F32x8 acc[4];
    #pragma unroll
    for (int t = 0; t < 4; ++t)
        #pragma unroll
        for (int r = 0; r < 8; ++r) acc[t].f[r] = 0.0f;

    const int mlo = lane & 15;
    const int hi  = lane >> 4;

    // ---- staging coordinates (A: 2 chunks/thread, B: 1 chunk/thread) -------
    // A tile 128x32 halves = 512 v8h units; unit u -> row u>>2, k8 (u&3)*8
    size_t abase[2]; int alds[2];
    #pragma unroll
    for (int i = 0; i < 2; ++i) {
        int u  = tid + i * 256;
        int r  = u >> 2;
        int c8 = (u & 3) * 8;
        int gr = m0 + r; if (gr > M - 1) gr = M - 1;     // clamp: safe, rows >= M never stored
        abase[i] = (size_t)gr * K + c8;
        alds[i]  = r * LDA + c8;
    }
    // B tile 64 cols x 32 halves = 256 v8h units; unit u -> col u>>2, k8 (u&3)*8
    int u  = tid & 255;
    int bc = u >> 2;
    int k8 = (u & 3) * 8;
    int gc = n0 + bc; if (gc > N - 1) gc = N - 1;        // clamp: cols >= N never stored
    const size_t bbase = (size_t)gc * K + k8;
    const int    blds  = bc * LDBT + k8;

    // ---- prologue: prefetch first K-tile into registers ---------------------
    v8h a0 = *(const v8h*)&A [abase[0]];
    v8h a1 = *(const v8h*)&A [abase[1]];
    v8h b0 = *(const v8h*)&Wt[bbase];

    for (int k0 = 0; k0 < K; k0 += BK) {
        *(v8h*)&As [alds[0]] = a0;
        *(v8h*)&As [alds[1]] = a1;
        *(v8h*)&BsT[blds]    = b0;
        __syncthreads();

        const int kn = k0 + BK;
        if (kn < K) {                                    // prefetch next tile
            a0 = *(const v8h*)&A [abase[0] + kn];
            a1 = *(const v8h*)&A [abase[1] + kn];
            b0 = *(const v8h*)&Wt[bbase    + kn];
        }

        // A fragment: lane l -> M = l&15 ; K groups of 8 chosen by l>>4
        F16x16 afrag;
        const int arow = wave * 16 + mlo;
        afrag.h8[0] = *(const v8h*)&As[arow * LDA + hi * 8];
        afrag.h8[1] = *(const v8h*)&As[arow * LDA + 16 + hi * 8];

        #pragma unroll
        for (int t = 0; t < 4; ++t) {
            F16x16 bfrag;
            const int bcol = t * 16 + mlo;
            bfrag.h8[0] = *(const v8h*)&BsT[bcol * LDBT + hi * 8];
            bfrag.h8[1] = *(const v8h*)&BsT[bcol * LDBT + 16 + hi * 8];
            acc[t].v = __builtin_amdgcn_wmma_f32_16x16x32_f16(
                false, afrag.v, false, bfrag.v, (short)0, acc[t].v, false, false);
        }
        __syncthreads();
    }

    // D layout: VGPR r, lane l -> M = r + 8*(l>>4), N = l&15
    #pragma unroll
    for (int t = 0; t < 4; ++t) {
        int col = n0 + t * 16 + mlo;
        if (col >= N) continue;
        #pragma unroll
        for (int r = 0; r < 8; ++r) {
            int row = m0 + wave * 16 + hi * 8 + r;
            if (row < M) G[(size_t)row * N + col] = acc[t].f[r];
        }
    }
}

// ---------------------------------------------------------------------------
// Conversion helpers
// ---------------------------------------------------------------------------
__global__ void f32_to_f16(const float* __restrict__ src, _Float16* __restrict__ dst,
                           size_t n)
{
    size_t t = (size_t)blockIdx.x * blockDim.x + threadIdx.x;
    if (t < n) dst[t] = (_Float16)src[t];
}

// Wt[n*K + k] = (f16) W[k*N + n]   (coalesced writes, strided reads; W is small)
__global__ void convert_w_t(const float* __restrict__ W, _Float16* __restrict__ Wt,
                            int K, int N)
{
    int t = blockIdx.x * blockDim.x + threadIdx.x;
    if (t >= K * N) return;
    int n = t / K, k = t % K;
    Wt[t] = (_Float16)W[(size_t)k * N + n];
}

// ---------------------------------------------------------------------------
// Attention / segment-softmax / scatter kernels (memory bound, VALU)
// ---------------------------------------------------------------------------
__device__ __forceinline__ float leaky02(float v) { return v > 0.f ? v : 0.2f * v; }

__device__ __forceinline__ void atomicMaxFloat(float* addr, float value) {
    if (value >= 0.f) atomicMax((int*)addr, __float_as_int(value));
    else              atomicMin((unsigned int*)addr, __float_as_uint(value));
}

__device__ __forceinline__ void edge_endpoints(const long long* __restrict__ ei,
                                               int e, int& s, int& d) {
    if (e < N_EDGES) { s = (int)ei[e]; d = (int)ei[N_EDGES + e]; }
    else             { s = d = e - N_EDGES; }           // self loop
}

__global__ void node_scores(const float* __restrict__ G,
                            const float* __restrict__ a_src,
                            const float* __restrict__ a_dst,
                            float* __restrict__ es, float* __restrict__ ed, int C)
{
    int t = blockIdx.x * blockDim.x + threadIdx.x;
    if (t >= N_NODES * HEADS) return;
    int node = t / HEADS, h = t % HEADS;
    const float* gp = G + (size_t)node * HEADS * C + (size_t)h * C;
    const float* as = a_src + h * C;
    const float* ad = a_dst + h * C;
    float s0 = 0.f, s1 = 0.f;
    for (int c = 0; c < C; ++c) { float g = gp[c]; s0 += g * as[c]; s1 += g * ad[c]; }
    es[t] = s0; ed[t] = s1;
}

__global__ void init_mden(float* __restrict__ m, float* __restrict__ den, int n)
{
    int t = blockIdx.x * blockDim.x + threadIdx.x;
    if (t < n) { m[t] = -1e30f; den[t] = 0.f; }
}

__global__ void zero_f32(float* __restrict__ p, size_t n)
{
    size_t t = (size_t)blockIdx.x * blockDim.x + threadIdx.x;
    if (t < n) p[t] = 0.f;
}

__global__ void edge_max(const long long* __restrict__ ei,
                         const float* __restrict__ es, const float* __restrict__ ed,
                         float* __restrict__ m)
{
    int t = blockIdx.x * blockDim.x + threadIdx.x;
    if (t >= E_TOT * HEADS) return;
    int e = t / HEADS, h = t % HEADS, s, d;
    edge_endpoints(ei, e, s, d);
    float v = leaky02(es[s * HEADS + h] + ed[d * HEADS + h]);
    atomicMaxFloat(&m[d * HEADS + h], v);
}

__global__ void edge_exp(const long long* __restrict__ ei,
                         const float* __restrict__ es, const float* __restrict__ ed,
                         const float* __restrict__ m,
                         float* __restrict__ ex, float* __restrict__ den)
{
    int t = blockIdx.x * blockDim.x + threadIdx.x;
    if (t >= E_TOT * HEADS) return;
    int e = t / HEADS, h = t % HEADS, s, d;
    edge_endpoints(ei, e, s, d);
    float v  = leaky02(es[s * HEADS + h] + ed[d * HEADS + h]);
    float xv = __expf(v - m[d * HEADS + h]);
    ex[t] = xv;
    atomicAdd(&den[d * HEADS + h], xv);
}

__global__ void edge_scatter(const long long* __restrict__ ei,
                             const float* __restrict__ G,
                             const float* __restrict__ ex, const float* __restrict__ den,
                             float* __restrict__ out, int HC, int C)
{
    size_t t = (size_t)blockIdx.x * blockDim.x + threadIdx.x;
    size_t total = (size_t)E_TOT * HC;
    if (t >= total) return;
    int e = (int)(t / HC);
    int j = (int)(t % HC);
    int h = j / C, s, d;
    edge_endpoints(ei, e, s, d);
    float alpha = ex[(size_t)e * HEADS + h] / (den[d * HEADS + h] + 1e-16f);
    atomicAdd(&out[(size_t)d * HC + j], G[(size_t)s * HC + j] * alpha);
}

// act: 0 = relu, 1 = leaky_relu(0.1); writes f16 activations for next layer
__global__ void finalize(const float* __restrict__ out, const float* __restrict__ bias,
                         _Float16* __restrict__ hnew, int HC, int act)
{
    int t = blockIdx.x * blockDim.x + threadIdx.x;
    if (t >= N_NODES * HC) return;
    int j = t % HC;
    float v = out[t] + bias[j];
    v = (act == 0) ? fmaxf(v, 0.f) : (v > 0.f ? v : 0.1f * v);
    hnew[t] = (_Float16)v;
}

__global__ void head_mean(const float* __restrict__ out, const float* __restrict__ bias,
                          float* __restrict__ y)
{
    int n = blockIdx.x * blockDim.x + threadIdx.x;
    if (n >= N_NODES) return;
    float s = 0.f;
    #pragma unroll
    for (int h = 0; h < HEADS; ++h) s += out[n * HEADS + h];
    y[n] = s * (1.0f / HEADS) + bias[0];
}

// ---------------------------------------------------------------------------
extern "C" void kernel_launch(void* const* d_in, const int* in_sizes, int n_in,
                              void* d_out, int out_size, void* d_ws, size_t ws_size,
                              hipStream_t stream)
{
    (void)in_sizes; (void)n_in; (void)out_size; (void)ws_size;

    const float*     x  = (const float*)d_in[0];
    const long long* ei = (const long long*)d_in[1];   // int64 edge_index [2,E]

    char*  ws   = (char*)d_ws;
    size_t offs = 0;
    auto alloc = [&](size_t bytes) {
        void* p = ws + offs;
        offs += (bytes + 255) & ~(size_t)255;
        return p;
    };
    _Float16* h16  = (_Float16*)alloc((size_t)N_NODES * 512 * 2);
    _Float16* wt16 = (_Float16*)alloc((size_t)512 * 512 * 2);
    float*    g    = (float*)alloc((size_t)N_NODES * 512 * 4);
    float*    outb = (float*)alloc((size_t)N_NODES * 512 * 4);
    float*    es   = (float*)alloc((size_t)N_NODES * HEADS * 4);
    float*    ed   = (float*)alloc((size_t)N_NODES * HEADS * 4);
    float*    m    = (float*)alloc((size_t)N_NODES * HEADS * 4);
    float*    den  = (float*)alloc((size_t)N_NODES * HEADS * 4);
    float*    ex   = (float*)alloc((size_t)E_TOT   * HEADS * 4);

    // layer-0 activations: x -> f16
    {
        size_t n = (size_t)N_NODES * 64;
        f32_to_f16<<<(unsigned)((n + 255) / 256), 256, 0, stream>>>(x, h16, n);
    }

    const int din[6] = {64, 64, 64, 256, 256, 512};
    const int Cs[6]  = {8, 8, 32, 32, 64, 1};
    const int act[6] = {0, 1, 0, 1, 0, 2};

    for (int L = 0; L < 6; ++L) {
        const float* Wp   = (const float*)d_in[2 + 4 * L + 0];
        const float* a_s  = (const float*)d_in[2 + 4 * L + 1];
        const float* a_d  = (const float*)d_in[2 + 4 * L + 2];
        const float* bias = (const float*)d_in[2 + 4 * L + 3];
        const int K = din[L], C = Cs[L], HC = HEADS * C;

        convert_w_t<<<(K * HC + 255) / 256, 256, 0, stream>>>(Wp, wt16, K, HC);

        dim3 gg((HC + BN - 1) / BN, (N_NODES + BM - 1) / BM);
        gemm_f16_wmma<<<gg, 256, 0, stream>>>(h16, wt16, g, N_NODES, K, HC);

        const int nh = N_NODES * HEADS;
        node_scores<<<(nh + 255) / 256, 256, 0, stream>>>(g, a_s, a_d, es, ed, C);
        init_mden<<<(nh + 255) / 256, 256, 0, stream>>>(m, den, nh);

        const size_t outn = (size_t)N_NODES * HC;
        zero_f32<<<(unsigned)((outn + 255) / 256), 256, 0, stream>>>(outb, outn);

        const int eh = E_TOT * HEADS;
        edge_max<<<(eh + 255) / 256, 256, 0, stream>>>(ei, es, ed, m);
        edge_exp<<<(eh + 255) / 256, 256, 0, stream>>>(ei, es, ed, m, ex, den);

        const size_t sc = (size_t)E_TOT * HC;
        edge_scatter<<<(unsigned)((sc + 255) / 256), 256, 0, stream>>>(ei, g, ex, den,
                                                                       outb, HC, C);
        if (act[L] != 2)
            finalize<<<(unsigned)((outn + 255) / 256), 256, 0, stream>>>(outb, bias, h16,
                                                                         HC, act[L]);
        else
            head_mean<<<(N_NODES + 255) / 256, 256, 0, stream>>>(outb, bias,
                                                                 (float*)d_out);
    }
}
